// SequenceTower_17420387353221
// MI455X (gfx1250) — compile-verified, matrix operand order checked by reading, not run on hardware
//
#include <hip/hip_runtime.h>
#include <hip/hip_bf16.h>

typedef _Float16 half_t;
typedef __attribute__((ext_vector_type(16))) _Float16 v16h;
typedef __attribute__((ext_vector_type(8)))  _Float16 v8h;
typedef __attribute__((ext_vector_type(8)))  float    v8f;

#define L_SEQ 8192
#define D_IN  128
#define D_HID 1024
#define D_FF  2048
#define NL    6
#define CHUNK 128
#define NCH   (L_SEQ / CHUNK)

__device__ __forceinline__ float sigmoidf_(float x) {
    return 1.0f / (1.0f + __expf(-x));
}

// ---------------------------------------------------------------------------
// fp32 -> f16 transposed weight conversion: W[K,N] row-major -> Wt[N,K]
// ---------------------------------------------------------------------------
__global__ void cvt_transpose(const float* __restrict__ W, half_t* __restrict__ Wt,
                              int K, int N) {
    size_t i = (size_t)blockIdx.x * 256 + threadIdx.x;
    size_t total = (size_t)K * N;
    if (i >= total) return;
    int n = (int)(i / K);
    int k = (int)(i % K);
    Wt[i] = (half_t)W[(size_t)k * N + n];
}

__global__ void cvt_plain(const float* __restrict__ x, half_t* __restrict__ y, size_t n) {
    size_t i = (size_t)blockIdx.x * 256 + threadIdx.x;
    if (i < n) y[i] = (half_t)x[i];
}

// ---------------------------------------------------------------------------
// WMMA GEMM: out[M,N] = epi( A[M,K](f16) @ Bt[N,K]^T (f16) + bias[N] )
// EPI: 0 = fp32 out, 1 = sigmoid fp32 out, 2 = residual add into outF (fp32),
//      3 = silu -> f16 out
// Block tile 128x128, BK=64 (2 WMMA k-steps per stage).
// 8 waves: 4 (m) x 2 (n); wave tile 32x64 -> 8 accumulators.
// Global->LDS via CDNA5 async-to-LDS DMA (ASYNCcnt), double-buffered,
// software-pipelined one K-tile ahead.
// ---------------------------------------------------------------------------
template <int EPI>
__global__ __launch_bounds__(256)
void gemm_wmma(const half_t* __restrict__ A, const half_t* __restrict__ Bt,
               const float* __restrict__ bias,
               float* __restrict__ outF, half_t* __restrict__ outH,
               int M, int N, int K) {
    // padded row stride 72 halves (144B, 16B-aligned)
    __shared__ half_t As[2][128][72];
    __shared__ half_t Bs[2][128][72];

    const int tid    = threadIdx.x;
    const int lane   = tid & 31;
    const int wid    = tid >> 5;
    const int wm     = wid & 3;      // wave row group  (4 x 32 rows)
    const int wn     = wid >> 2;     // wave col group  (2 x 64 cols)
    const int lane16 = lane & 15;
    const int hi     = lane >> 4;    // 0: K 0-7/16-23, 1: K 8-15/24-31

    const int m0 = blockIdx.y * 128;
    const int n0 = blockIdx.x * 128;

    v8f acc[2][4];
#pragma unroll
    for (int mi = 0; mi < 2; ++mi)
#pragma unroll
        for (int ni = 0; ni < 4; ++ni)
#pragma unroll
            for (int j = 0; j < 8; ++j) acc[mi][ni][j] = 0.0f;

    // async tile copy: 128x64 halves = 16KB per matrix; 256 threads x 4 B128 each
    const int ar  = tid >> 3;          // 0..31
    const int ac8 = (tid & 7) * 8;     // 0,8,...,56 (halves)

    auto issue_tile = [&](int buf, int kk) {
#pragma unroll
        for (int j = 0; j < 4; ++j) {
            const int row = ar + 32 * j;
            uint32_t da = (uint32_t)(uintptr_t)&As[buf][row][ac8];
            uint64_t ga = (uint64_t)(uintptr_t)(A + (size_t)(m0 + row) * K + kk + ac8);
            asm volatile("global_load_async_to_lds_b128 %0, %1, off"
                         :: "v"(da), "v"(ga) : "memory");
            uint32_t db = (uint32_t)(uintptr_t)&Bs[buf][row][ac8];
            uint64_t gb = (uint64_t)(uintptr_t)(Bt + (size_t)(n0 + row) * K + kk + ac8);
            asm volatile("global_load_async_to_lds_b128 %0, %1, off"
                         :: "v"(db), "v"(gb) : "memory");
        }
    };

    issue_tile(0, 0);

    int buf = 0;
    for (int kk = 0; kk < K; kk += 64) {
        if (kk + 64 < K) {
            issue_tile(buf ^ 1, kk + 64);            // next tile in flight
            asm volatile("s_wait_asynccnt 0x8" ::: "memory");  // current tile landed
        } else {
            asm volatile("s_wait_asynccnt 0x0" ::: "memory");
        }
        __syncthreads();

#pragma unroll
        for (int ks = 0; ks < 64; ks += 32) {
            v16h afrag[2];
#pragma unroll
            for (int mi = 0; mi < 2; ++mi) {
                int row = wm * 32 + mi * 16 + lane16;
                v8h lo = *(const v8h*)&As[buf][row][ks + hi * 8];
                v8h hv = *(const v8h*)&As[buf][row][ks + hi * 8 + 16];
#pragma unroll
                for (int j = 0; j < 8; ++j) { afrag[mi][j] = lo[j]; afrag[mi][j + 8] = hv[j]; }
            }
            v16h bfrag[4];
#pragma unroll
            for (int ni = 0; ni < 4; ++ni) {
                int row = wn * 64 + ni * 16 + lane16;
                v8h lo = *(const v8h*)&Bs[buf][row][ks + hi * 8];
                v8h hv = *(const v8h*)&Bs[buf][row][ks + hi * 8 + 16];
#pragma unroll
                for (int j = 0; j < 8; ++j) { bfrag[ni][j] = lo[j]; bfrag[ni][j + 8] = hv[j]; }
            }
#pragma unroll
            for (int mi = 0; mi < 2; ++mi)
#pragma unroll
                for (int ni = 0; ni < 4; ++ni)
                    acc[mi][ni] = __builtin_amdgcn_wmma_f32_16x16x32_f16(
                        false, afrag[mi], false, bfrag[ni],
                        (short)0, acc[mi][ni], false, false);
        }
        __syncthreads();   // all reads of buf done before it is refilled
        buf ^= 1;
    }

    // -------- epilogue ----------------------------------------------------
#pragma unroll
    for (int mi = 0; mi < 2; ++mi) {
#pragma unroll
        for (int ni = 0; ni < 4; ++ni) {
            int gn  = n0 + wn * 64 + ni * 16 + lane16;
            float bv = bias[gn];
            int gm0 = m0 + wm * 32 + mi * 16 + hi * 8;
#pragma unroll
            for (int r = 0; r < 8; ++r) {
                size_t idx = (size_t)(gm0 + r) * N + gn;
                float v = acc[mi][ni][r] + bv;
                if (EPI == 0) {
                    outF[idx] = v;
                } else if (EPI == 1) {
                    outF[idx] = sigmoidf_(v);
                } else if (EPI == 2) {
                    outF[idx] = outF[idx] + v;
                } else {
                    outH[idx] = (half_t)(v * sigmoidf_(v));
                }
            }
        }
    }
}

// ---------------------------------------------------------------------------
// LayerNorm over D_HID. One block per row. OUTH: write f16, else fp32.
// ---------------------------------------------------------------------------
template <bool OUTH>
__global__ __launch_bounds__(256)
void ln_kernel(const float* __restrict__ x, const float* __restrict__ g,
               const float* __restrict__ b, float* __restrict__ of,
               half_t* __restrict__ oh) {
    __shared__ float s1[256], s2[256];
    const int tid = threadIdx.x;
    const size_t row = blockIdx.x;
    const float* xr = x + row * D_HID;
    float v[4], sum = 0.f, sq = 0.f;
#pragma unroll
    for (int j = 0; j < 4; ++j) {
        v[j] = xr[tid + j * 256];
        sum += v[j]; sq += v[j] * v[j];
    }
    s1[tid] = sum; s2[tid] = sq;
    __syncthreads();
    for (int off = 128; off > 0; off >>= 1) {
        if (tid < off) { s1[tid] += s1[tid + off]; s2[tid] += s2[tid + off]; }
        __syncthreads();
    }
    float mean = s1[0] * (1.0f / D_HID);
    float var  = s2[0] * (1.0f / D_HID) - mean * mean;
    float rstd = rsqrtf(var + 1e-5f);
#pragma unroll
    for (int j = 0; j < 4; ++j) {
        int c = tid + j * 256;
        float y = (v[j] - mean) * rstd * g[c] + b[c];
        if (OUTH) oh[row * D_HID + c] = (half_t)y;
        else      of[row * D_HID + c] = y;
    }
}

// ---------------------------------------------------------------------------
// Chunked channelwise linear scan: b_t = u_t + a * b_{t-1}
// ---------------------------------------------------------------------------
template <bool FWD>
__global__ __launch_bounds__(256)
void scan_chunk(const float* __restrict__ u, const float* __restrict__ dp,
                float* __restrict__ out, float* __restrict__ carry) {
    const int c = blockIdx.y * 256 + threadIdx.x;
    const int chunk = blockIdx.x;
    const float a = sigmoidf_(dp[c]);
    const int base = chunk * CHUNK;
    float bacc = 0.f;
    for (int i = 0; i < CHUNK; ++i) {
        int t = FWD ? (base + i) : (base + CHUNK - 1 - i);
        size_t idx = (size_t)t * D_HID + c;
        bacc = u[idx] + a * bacc;
        out[idx] = bacc;
    }
    carry[(size_t)chunk * D_HID + c] = bacc;
}

template <bool FWD>
__global__ __launch_bounds__(256)
void scan_carry(float* __restrict__ carry, const float* __restrict__ dp) {
    const int c = blockIdx.x * 256 + threadIdx.x;
    const float a = sigmoidf_(dp[c]);
    float aC = a;
#pragma unroll
    for (int j = 0; j < 7; ++j) aC *= aC;   // a^128
    float pref = 0.f;
    for (int k = 0; k < NCH; ++k) {
        int kk = FWD ? k : (NCH - 1 - k);
        size_t idx = (size_t)kk * D_HID + c;
        float tmp = carry[idx];
        carry[idx] = pref;                  // exclusive carry-in for chunk kk
        pref = tmp + aC * pref;
    }
}

template <bool FWD>
__global__ __launch_bounds__(256)
void scan_apply(float* __restrict__ out, const float* __restrict__ carry,
                const float* __restrict__ dp) {
    const int c = blockIdx.y * 256 + threadIdx.x;
    const int chunk = blockIdx.x;
    const float a = sigmoidf_(dp[c]);
    const float cin = carry[(size_t)chunk * D_HID + c];
    const int base = chunk * CHUNK;
    float ap = a;
    for (int i = 0; i < CHUNK; ++i) {
        int t = FWD ? (base + i) : (base + CHUNK - 1 - i);
        out[(size_t)t * D_HID + c] += ap * cin;
        ap *= a;
    }
}

// state = 0.5 * (fwd + bwd) * gate  ->  f16
__global__ __launch_bounds__(256)
void state_kernel(const float* __restrict__ f, const float* __restrict__ bwd,
                  const float* __restrict__ gate, half_t* __restrict__ out, size_t n) {
    size_t i = (size_t)blockIdx.x * 256 + threadIdx.x;
    if (i < n) out[i] = (half_t)(0.5f * (f[i] + bwd[i]) * gate[i]);
}

// ---------------------------------------------------------------------------
extern "C" void kernel_launch(void* const* d_in, const int* in_sizes, int n_in,
                              void* d_out, int out_size, void* d_ws, size_t ws_size,
                              hipStream_t stream) {
    const float* nf      = (const float*)d_in[0];
    const float* w_in    = (const float*)d_in[1];
    const float* b_in    = (const float*)d_in[2];
    const float* ln1_g   = (const float*)d_in[3];
    const float* ln1_b   = (const float*)d_in[4];
    const float* w_u     = (const float*)d_in[5];
    const float* b_u     = (const float*)d_in[6];
    const float* w_g     = (const float*)d_in[7];
    const float* b_g     = (const float*)d_in[8];
    const float* w_o     = (const float*)d_in[9];
    const float* b_o     = (const float*)d_in[10];
    const float* decay_p = (const float*)d_in[11];
    const float* ln2_g   = (const float*)d_in[12];
    const float* ln2_b   = (const float*)d_in[13];
    const float* w_f1    = (const float*)d_in[14];
    const float* b_f1    = (const float*)d_in[15];
    const float* w_f2    = (const float*)d_in[16];
    const float* b_f2    = (const float*)d_in[17];
    const float* lnf_g   = (const float*)d_in[18];
    const float* lnf_b   = (const float*)d_in[19];
    float* out = (float*)d_out;

    // ---- workspace bump allocator (256B aligned) ------------------------
    char* wp = (char*)d_ws;
    auto alloc = [&](size_t bytes) -> void* {
        void* p = (void*)wp;
        wp += (bytes + 255) & ~(size_t)255;
        return p;
    };
    const size_t LD = (size_t)L_SEQ * D_HID;
    float*  xf32  = (float*) alloc(LD * 4);
    float*  u     = (float*) alloc(LD * 4);
    float*  fwdb  = (float*) alloc(LD * 4);
    float*  bwdb  = (float*) alloc(LD * 4);
    float*  gate  = (float*) alloc(LD * 4);
    half_t* hf16  = (half_t*)alloc(LD * 2);
    half_t* yf16  = (half_t*)alloc((size_t)L_SEQ * D_FF * 2);
    half_t* nf16  = (half_t*)alloc((size_t)L_SEQ * D_IN * 2);
    half_t* winT  = (half_t*)alloc((size_t)D_HID * D_IN * 2);
    half_t* wuT[NL];  half_t* wgT[NL];  half_t* woT[NL];
    half_t* wf1T[NL]; half_t* wf2T[NL];
    for (int l = 0; l < NL; ++l) {
        wuT[l]  = (half_t*)alloc((size_t)D_HID * D_HID * 2);
        wgT[l]  = (half_t*)alloc((size_t)D_HID * D_HID * 2);
        woT[l]  = (half_t*)alloc((size_t)D_HID * D_HID * 2);
        wf1T[l] = (half_t*)alloc((size_t)D_FF * D_HID * 2);
        wf2T[l] = (half_t*)alloc((size_t)D_HID * D_FF * 2);
    }
    float* carryF = (float*)alloc((size_t)NCH * D_HID * 4);
    float* carryB = (float*)alloc((size_t)NCH * D_HID * 4);

    // ---- weight / input conversion --------------------------------------
    {
        size_t n = (size_t)L_SEQ * D_IN;
        cvt_plain<<<(unsigned)((n + 255) / 256), 256, 0, stream>>>(nf, nf16, n);
        size_t m = (size_t)D_IN * D_HID;
        cvt_transpose<<<(unsigned)((m + 255) / 256), 256, 0, stream>>>(w_in, winT, D_IN, D_HID);
        size_t sq = (size_t)D_HID * D_HID, ff = (size_t)D_HID * D_FF;
        for (int l = 0; l < NL; ++l) {
            cvt_transpose<<<(unsigned)((sq + 255) / 256), 256, 0, stream>>>(w_u  + l * sq, wuT[l],  D_HID, D_HID);
            cvt_transpose<<<(unsigned)((sq + 255) / 256), 256, 0, stream>>>(w_g  + l * sq, wgT[l],  D_HID, D_HID);
            cvt_transpose<<<(unsigned)((sq + 255) / 256), 256, 0, stream>>>(w_o  + l * sq, woT[l],  D_HID, D_HID);
            cvt_transpose<<<(unsigned)((ff + 255) / 256), 256, 0, stream>>>(w_f1 + l * ff, wf1T[l], D_HID, D_FF);
            cvt_transpose<<<(unsigned)((ff + 255) / 256), 256, 0, stream>>>(w_f2 + l * ff, wf2T[l], D_FF, D_HID);
        }
    }

    const dim3 blk(256);
    const dim3 gD (D_HID / 128, L_SEQ / 128);   // N=1024 GEMMs
    const dim3 gFF(D_FF  / 128, L_SEQ / 128);   // N=2048 GEMM
    const dim3 gScan(NCH, D_HID / 256);
    const unsigned nElemBlocks = (unsigned)((LD + 255) / 256);

    // ---- input projection: x = nf @ w_in + b_in --------------------------
    gemm_wmma<0><<<gD, blk, 0, stream>>>(nf16, winT, b_in, xf32, nullptr,
                                         L_SEQ, D_HID, D_IN);

    // ---- layers ----------------------------------------------------------
    for (int l = 0; l < NL; ++l) {
        const float* dp = decay_p + (size_t)l * D_HID;

        // h = LN1(x) -> f16
        ln_kernel<true><<<L_SEQ, blk, 0, stream>>>(xf32, ln1_g + l * D_HID,
                                                   ln1_b + l * D_HID, nullptr, hf16);
        // u = h @ wu + bu ; gate = sigmoid(h @ wg + bg)
        gemm_wmma<0><<<gD, blk, 0, stream>>>(hf16, wuT[l], b_u + l * D_HID,
                                             u, nullptr, L_SEQ, D_HID, D_HID);
        gemm_wmma<1><<<gD, blk, 0, stream>>>(hf16, wgT[l], b_g + l * D_HID,
                                             gate, nullptr, L_SEQ, D_HID, D_HID);
        // bidirectional chunked scans
        scan_chunk<true ><<<gScan, blk, 0, stream>>>(u, dp, fwdb, carryF);
        scan_chunk<false><<<gScan, blk, 0, stream>>>(u, dp, bwdb, carryB);
        scan_carry<true ><<<D_HID / 256, blk, 0, stream>>>(carryF, dp);
        scan_carry<false><<<D_HID / 256, blk, 0, stream>>>(carryB, dp);
        scan_apply<true ><<<gScan, blk, 0, stream>>>(fwdb, carryF, dp);
        scan_apply<false><<<gScan, blk, 0, stream>>>(bwdb, carryB, dp);
        // state = 0.5*(fwd+bwd)*gate -> f16 (reuse hf16)
        state_kernel<<<nElemBlocks, blk, 0, stream>>>(fwdb, bwdb, gate, hf16, LD);
        // x += state @ wo + bo
        gemm_wmma<2><<<gD, blk, 0, stream>>>(hf16, woT[l], b_o + l * D_HID,
                                             xf32, nullptr, L_SEQ, D_HID, D_HID);
        // FFN: t = LN2(x); y = silu(t@wf1+bf1); x += y@wf2+bf2
        ln_kernel<true><<<L_SEQ, blk, 0, stream>>>(xf32, ln2_g + l * D_HID,
                                                   ln2_b + l * D_HID, nullptr, hf16);
        gemm_wmma<3><<<gFF, blk, 0, stream>>>(hf16, wf1T[l], b_f1 + (size_t)l * D_FF,
                                              nullptr, yf16, L_SEQ, D_FF, D_HID);
        gemm_wmma<2><<<gD, blk, 0, stream>>>(yf16, wf2T[l], b_f2 + l * D_HID,
                                             xf32, nullptr, L_SEQ, D_HID, D_FF);
    }

    // ---- final LN -> fp32 output ----------------------------------------
    ln_kernel<false><<<L_SEQ, blk, 0, stream>>>(xf32, lnf_g, lnf_b, out, nullptr);
}
